// SelfAttention_18193481465940
// MI455X (gfx1250) — matrix operand order repaired and checked by reading, not compiled, not run on hardware
//
#include <hip/hip_runtime.h>

// ---------------------------------------------------------------------------
// Flash-attention  B=2, S=2048, D=1024, H=16 (head_dim=64), fp32 in/out.
// CDNA5 (gfx1250): v_wmma_f32_16x16x32_bf16 + async Tensor Data Mover staging
// with double-buffered LDS tiles (TDM of block n+1 overlaps compute of n).
//
// Pass A (once per call): K -> bf16 [B,S,D];  V -> bf16 transposed [B,H,64,S]
// Pass B: flash attention; K/V 64x64 tiles staged into LDS by TDM
//         (tensor_load_to_lds with hardware pad -> LDK=72 layout),
//         QK^T and PV on WMMA bf16, online softmax in f32 registers.
// Fallback (ws too small): cooperative-load kernel.
// ---------------------------------------------------------------------------

typedef __attribute__((ext_vector_type(16))) __bf16 v16bf;
typedef __attribute__((ext_vector_type(8)))  float  v8f;
typedef __attribute__((ext_vector_type(4)))  unsigned int u32x4;
typedef __attribute__((ext_vector_type(8)))  int    i32x8;
typedef __attribute__((ext_vector_type(4)))  int    i32x4;

#define S_LEN   2048
#define D_MODEL 1024
#define N_HEADS 16
#define DH      64          // head dim
#define KC      64          // keys per block
#define NBLK    (S_LEN / KC)
#define LDK     72          // LDS row stride in elements (144B = 128B data + 16B pad)

union FragBF {
    v16bf v;
    uint4 q[2];
    unsigned short u[16];
};

__device__ __forceinline__ unsigned short f2bf(float f) {
    unsigned int u = __float_as_uint(f);
    unsigned int r = u + 0x7FFFu + ((u >> 16) & 1u);   // round-to-nearest-even
    return (unsigned short)(r >> 16);
}

// --------------------------- TDM tile descriptor ---------------------------
// 2D tile: 64 rows x 64 bf16 elements, row stride `stride_el` elements in
// global memory, destination LDS rows padded 128B->144B (matches LDK=72).
__device__ __forceinline__ void tdm_load_tile64x64_bf16(const unsigned short* gsrc,
                                                        unsigned int lds_byte_off,
                                                        unsigned int stride_el) {
    const unsigned long long ga = (unsigned long long)(uintptr_t)gsrc;
    u32x4 g0;
    g0[0] = 1u;                                            // count=1 (valid user D#)
    g0[1] = lds_byte_off;                                  // lds_addr
    g0[2] = (unsigned int)(ga & 0xFFFFFFFFull);            // global_addr[31:0]
    g0[3] = (unsigned int)((ga >> 32) & 0x01FFFFFFull)     // global_addr[56:32]
          | (2u << 30);                                    // type = 2 ("image")
    i32x8 g1;
    g1[0] = (1 << 16)     // data_size = 1 -> 2 bytes
          | (1 << 20)     // pad_enable
          | (4 << 22)     // pad_interval: every 32 DWORDs (128B)
          | (3 << 25);    // pad_amount: 4 DWORDs (16B)
    g1[1] = (64 << 16);   // tensor_dim0[15:0] = 64
    g1[2] = (64 << 16);   // tensor_dim0[31:16]=0 | tensor_dim1[15:0] = 64
    g1[3] = (64 << 16);   // tensor_dim1[31:16]=0 | tile_dim0 = 64
    g1[4] = 64;           // tile_dim1 = 64, tile_dim2 = 0
    g1[5] = (int)stride_el; // tensor_dim0_stride[31:0]
    g1[6] = 0;            // stride hi / tensor_dim1_stride lo (unused, 2D)
    g1[7] = 0;
    const i32x4 z4 = {0, 0, 0, 0};
    const i32x8 z8 = {0, 0, 0, 0, 0, 0, 0, 0};
    __builtin_amdgcn_tensor_load_to_lds(g0, g1, z4, z4, z8, 0);
}

// --------------------------- Pass A kernels --------------------------------
__global__ __launch_bounds__(256)
void convert_bf16_kernel(const float* __restrict__ src, unsigned short* __restrict__ dst) {
    const size_t idx = ((size_t)blockIdx.x * blockDim.x + threadIdx.x) * 8;
    const float4 a = *(const float4*)(src + idx);
    const float4 b = *(const float4*)(src + idx + 4);
    ushort4 o0, o1;
    o0.x = f2bf(a.x); o0.y = f2bf(a.y); o0.z = f2bf(a.z); o0.w = f2bf(a.w);
    o1.x = f2bf(b.x); o1.y = f2bf(b.y); o1.z = f2bf(b.z); o1.w = f2bf(b.w);
    *(ushort4*)(dst + idx)     = o0;
    *(ushort4*)(dst + idx + 4) = o1;
}

// V [B,S,D] f32  ->  Vt [B,H,DH,S] bf16 (per-head transpose), 64x64 tiles.
__global__ __launch_bounds__(256)
void transpose_v_kernel(const float* __restrict__ Vg, unsigned short* __restrict__ Vt) {
    __shared__ unsigned short t[64 * 65];
    const int s0  = blockIdx.x * 64;
    const int h   = blockIdx.y;
    const int b   = blockIdx.z;
    const int tid = threadIdx.x;
    const int j4 = (tid & 15) * 4;   // dh base
    const int iq = tid >> 4;         // 0..15
    #pragma unroll
    for (int p = 0; p < 4; ++p) {
        const int i = p * 16 + iq;   // s row within tile
        const float4 v4 = *(const float4*)(Vg + ((size_t)b * S_LEN + s0 + i) * D_MODEL
                                              + h * DH + j4);
        t[i * 65 + j4 + 0] = f2bf(v4.x);
        t[i * 65 + j4 + 1] = f2bf(v4.y);
        t[i * 65 + j4 + 2] = f2bf(v4.z);
        t[i * 65 + j4 + 3] = f2bf(v4.w);
    }
    __syncthreads();
    const int j  = tid >> 2;         // dh row 0..63
    const int i0 = (tid & 3) * 16;   // s chunk
    unsigned short* orow = Vt + (((size_t)b * N_HEADS + h) * DH + j) * S_LEN + s0 + i0;
    #pragma unroll
    for (int k = 0; k < 16; ++k) orow[k] = t[(i0 + k) * 65 + j];
}

// ------------------- Pass B: double-buffered TDM + WMMA --------------------
__global__ __launch_bounds__(128)
void fattn_wmma_tdm(const float* __restrict__ Qg, const unsigned short* __restrict__ Kbf,
                    const unsigned short* __restrict__ Vtb, float* __restrict__ Og) {
    __shared__ __align__(16) unsigned short sK [2][KC * LDK];   // [buf][key][dh]
    __shared__ __align__(16) unsigned short sVT[2][DH * LDK];   // [buf][dh][key]
    __shared__ __align__(16) unsigned short sP [4 * 16 * LDK];

    const int tid   = threadIdx.x;
    const int wave  = tid >> 5;
    const int lane  = tid & 31;
    const int lhalf = lane >> 4;
    const int l16   = lane & 15;
    const int wid   = __builtin_amdgcn_readfirstlane(tid) >> 5;   // wave-uniform (SGPR)

    const int qtile = blockIdx.x;
    const int h     = blockIdx.y;
    const int b     = blockIdx.z;

    const int    q0      = qtile * 64 + wave * 16;
    const size_t base_bh = (size_t)b * S_LEN * D_MODEL + (size_t)h * DH;
    const unsigned short* Kbase = Kbf + base_bh;                                // [key][dh]
    const unsigned short* Vbase = Vtb + ((size_t)b * N_HEADS + h) * DH * S_LEN; // [dh][s]

    unsigned int ldsK[2], ldsV[2];
    ldsK[0] = (unsigned int)(uintptr_t)(void*)sK[0];
    ldsK[1] = (unsigned int)(uintptr_t)(void*)sK[1];
    ldsV[0] = (unsigned int)(uintptr_t)(void*)sVT[0];
    ldsV[1] = (unsigned int)(uintptr_t)(void*)sVT[1];

    // ---- Q fragments (A layout), pre-scaled by 1/sqrt(D) = 2^-5 (exact) ----
    FragBF qf[2];
    {
        const float* qrow = Qg + base_bh + (size_t)(q0 + l16) * D_MODEL;
        #pragma unroll
        for (int c = 0; c < 2; ++c) {
            const int cb = 32 * c + (lhalf ? 16 : 0);
            #pragma unroll
            for (int j = 0; j < 8; ++j) {
                qf[c].u[j]     = f2bf(qrow[cb + j]      * 0.03125f);
                qf[c].u[8 + j] = f2bf(qrow[cb + 8 + j] * 0.03125f);
            }
        }
    }

    v8f o[4];
    {
        v8f z = {};
        #pragma unroll
        for (int t = 0; t < 4; ++t) o[t] = z;
    }
    float m[8], lsum[8];
    #pragma unroll
    for (int r = 0; r < 8; ++r) { m[r] = -1e30f; lsum[r] = 0.0f; }

    unsigned short* pw = &sP[wave * 16 * LDK];

    // ---- prologue: async-stage block 0 into buffer 0 ----
    if (wid == 0) {
        tdm_load_tile64x64_bf16(Kbase, ldsK[0], D_MODEL);
    } else if (wid == 1) {
        tdm_load_tile64x64_bf16(Vbase, ldsV[0], S_LEN);
    }

    for (int blk = 0; blk < NBLK; ++blk) {
        const int cur = blk & 1;
        __builtin_amdgcn_s_wait_tensorcnt(0);  // buffer `cur` filled (waves 0/1)
        __syncthreads();                       // visible to all; prev reads of cur^1 done

        // ---- async TDM for the NEXT block overlaps this block's compute ----
        if (blk + 1 < NBLK) {
            const int kb1 = (blk + 1) * KC;
            if (wid == 0) {
                tdm_load_tile64x64_bf16(Kbase + (size_t)kb1 * D_MODEL, ldsK[cur ^ 1], D_MODEL);
            } else if (wid == 1) {
                tdm_load_tile64x64_bf16(Vbase + kb1, ldsV[cur ^ 1], S_LEN);
            } else if (wid == 2 && blk + 2 < NBLK) {
                const int kb2 = (blk + 2) * KC;
                __builtin_prefetch(Kbase + (size_t)(kb2 + (lane >> 2)) * D_MODEL
                                         + (lane & 3) * 16, 0, 1);
                __builtin_prefetch(Vbase + (size_t)(lane * 2) * S_LEN + kb2, 0, 1);
            }
        }

        const unsigned short* cK = sK[cur];
        const unsigned short* cV = sVT[cur];

        // ---- S = Q K^T : 4 key-tiles x 2 dh-chunks of WMMA ----
        v8f sc[4];
        #pragma unroll
        for (int nt = 0; nt < 4; ++nt) {
            FragBF b0, b1;
            {
                const unsigned short* p0 = &cK[(nt * 16 + l16) * LDK + 0  + 16 * lhalf];
                const unsigned short* p1 = &cK[(nt * 16 + l16) * LDK + 32 + 16 * lhalf];
                b0.q[0] = *(const uint4*)p0; b0.q[1] = *(const uint4*)(p0 + 8);
                b1.q[0] = *(const uint4*)p1; b1.q[1] = *(const uint4*)(p1 + 8);
            }
            v8f acc = {};
            acc = __builtin_amdgcn_wmma_f32_16x16x32_bf16(false, qf[0].v, false, b0.v,
                                                          (short)0, acc, false, false);
            acc = __builtin_amdgcn_wmma_f32_16x16x32_bf16(false, qf[1].v, false, b1.v,
                                                          (short)0, acc, false, false);
            sc[nt] = acc;
        }

        // ---- online softmax (rows r / r+8 across lane halves) ----
        #pragma unroll
        for (int r = 0; r < 8; ++r) {
            float bm = fmaxf(fmaxf(sc[0][r], sc[1][r]), fmaxf(sc[2][r], sc[3][r]));
            bm = fmaxf(bm, __shfl_xor(bm, 1));
            bm = fmaxf(bm, __shfl_xor(bm, 2));
            bm = fmaxf(bm, __shfl_xor(bm, 4));
            bm = fmaxf(bm, __shfl_xor(bm, 8));
            const float mn    = fmaxf(m[r], bm);
            const float alpha = __expf(m[r] - mn);
            m[r] = mn;
            float rs = 0.0f;
            #pragma unroll
            for (int nt = 0; nt < 4; ++nt) {
                const float pv = __expf(sc[nt][r] - mn);
                sc[nt][r] = pv;
                rs += pv;
            }
            rs += __shfl_xor(rs, 1);
            rs += __shfl_xor(rs, 2);
            rs += __shfl_xor(rs, 4);
            rs += __shfl_xor(rs, 8);
            lsum[r] = lsum[r] * alpha + rs;
            #pragma unroll
            for (int t = 0; t < 4; ++t) o[t][r] = o[t][r] * alpha;
        }

        // ---- C-layout P -> LDS (bf16), re-read in A layout ----
        #pragma unroll
        for (int r = 0; r < 8; ++r) {
            const int row = r + 8 * lhalf;
            #pragma unroll
            for (int nt = 0; nt < 4; ++nt)
                pw[row * LDK + nt * 16 + l16] = f2bf(sc[nt][r]);
        }
        __syncthreads();

        // ---- O += P V : 2 key-chunks x 4 dh-tiles of WMMA ----
        #pragma unroll
        for (int c = 0; c < 2; ++c) {
            FragBF pf;
            const unsigned short* pp = &pw[l16 * LDK + 32 * c + 16 * lhalf];
            pf.q[0] = *(const uint4*)pp;
            pf.q[1] = *(const uint4*)(pp + 8);
            #pragma unroll
            for (int vt = 0; vt < 4; ++vt) {
                FragBF vf;
                const unsigned short* vp = &cV[(vt * 16 + l16) * LDK + 32 * c + 16 * lhalf];
                vf.q[0] = *(const uint4*)vp; vf.q[1] = *(const uint4*)(vp + 8);
                o[vt] = __builtin_amdgcn_wmma_f32_16x16x32_bf16(false, pf.v, false, vf.v,
                                                                (short)0, o[vt], false, false);
            }
        }
    }

    #pragma unroll
    for (int r = 0; r < 8; ++r) {
        const float inv = 1.0f / lsum[r];
        const int   row = r + 8 * lhalf;
        float* orow = Og + base_bh + (size_t)(q0 + row) * D_MODEL;
        #pragma unroll
        for (int vt = 0; vt < 4; ++vt)
            orow[vt * 16 + l16] = o[vt][r] * inv;
    }
}

// --------------------------- Fallback kernel -------------------------------
__global__ __launch_bounds__(128)
void fattn_wmma_bf16_fb(const float* __restrict__ Qg, const float* __restrict__ Kg,
                        const float* __restrict__ Vg, float* __restrict__ Og) {
    __shared__ __align__(16) unsigned short sK [KC * LDK];
    __shared__ __align__(16) unsigned short sVT[DH * LDK];
    __shared__ __align__(16) unsigned short sP [4 * 16 * LDK];

    const int tid   = threadIdx.x;
    const int wave  = tid >> 5;
    const int lane  = tid & 31;
    const int lhalf = lane >> 4;
    const int l16   = lane & 15;
    const int qtile = blockIdx.x;
    const int h     = blockIdx.y;
    const int b     = blockIdx.z;
    const int    q0      = qtile * 64 + wave * 16;
    const size_t base_bh = (size_t)b * S_LEN * D_MODEL + (size_t)h * DH;

    FragBF qf[2];
    {
        const float* qrow = Qg + base_bh + (size_t)(q0 + l16) * D_MODEL;
        #pragma unroll
        for (int c = 0; c < 2; ++c) {
            const int cb = 32 * c + (lhalf ? 16 : 0);
            #pragma unroll
            for (int j = 0; j < 8; ++j) {
                qf[c].u[j]     = f2bf(qrow[cb + j]      * 0.03125f);
                qf[c].u[8 + j] = f2bf(qrow[cb + 8 + j] * 0.03125f);
            }
        }
    }
    v8f o[4];
    { v8f z = {}; for (int t = 0; t < 4; ++t) o[t] = z; }
    float m[8], lsum[8];
    #pragma unroll
    for (int r = 0; r < 8; ++r) { m[r] = -1e30f; lsum[r] = 0.0f; }
    unsigned short* pw = &sP[wave * 16 * LDK];

    for (int blk = 0; blk < NBLK; ++blk) {
        const int kb0 = blk * KC;
        __syncthreads();
        {
            const int rq = tid & 15;
            const int rb = tid >> 4;
            #pragma unroll
            for (int p = 0; p < 8; ++p) {
                const int key = p * 8 + rb;
                const size_t goff = base_bh + (size_t)(kb0 + key) * D_MODEL + rq * 4;
                const float4 k4 = *(const float4*)(Kg + goff);
                const float4 v4 = *(const float4*)(Vg + goff);
                ushort4 ks;
                ks.x = f2bf(k4.x); ks.y = f2bf(k4.y); ks.z = f2bf(k4.z); ks.w = f2bf(k4.w);
                *(ushort4*)&sK[key * LDK + rq * 4] = ks;
                sVT[(rq * 4 + 0) * LDK + key] = f2bf(v4.x);
                sVT[(rq * 4 + 1) * LDK + key] = f2bf(v4.y);
                sVT[(rq * 4 + 2) * LDK + key] = f2bf(v4.z);
                sVT[(rq * 4 + 3) * LDK + key] = f2bf(v4.w);
            }
        }
        __syncthreads();

        v8f sc[4];
        #pragma unroll
        for (int nt = 0; nt < 4; ++nt) {
            FragBF b0, b1;
            const unsigned short* p0 = &sK[(nt * 16 + l16) * LDK + 0  + 16 * lhalf];
            const unsigned short* p1 = &sK[(nt * 16 + l16) * LDK + 32 + 16 * lhalf];
            b0.q[0] = *(const uint4*)p0; b0.q[1] = *(const uint4*)(p0 + 8);
            b1.q[0] = *(const uint4*)p1; b1.q[1] = *(const uint4*)(p1 + 8);
            v8f acc = {};
            acc = __builtin_amdgcn_wmma_f32_16x16x32_bf16(false, qf[0].v, false, b0.v,
                                                          (short)0, acc, false, false);
            acc = __builtin_amdgcn_wmma_f32_16x16x32_bf16(false, qf[1].v, false, b1.v,
                                                          (short)0, acc, false, false);
            sc[nt] = acc;
        }
        #pragma unroll
        for (int r = 0; r < 8; ++r) {
            float bm = fmaxf(fmaxf(sc[0][r], sc[1][r]), fmaxf(sc[2][r], sc[3][r]));
            bm = fmaxf(bm, __shfl_xor(bm, 1));
            bm = fmaxf(bm, __shfl_xor(bm, 2));
            bm = fmaxf(bm, __shfl_xor(bm, 4));
            bm = fmaxf(bm, __shfl_xor(bm, 8));
            const float mn    = fmaxf(m[r], bm);
            const float alpha = __expf(m[r] - mn);
            m[r] = mn;
            float rs = 0.0f;
            #pragma unroll
            for (int nt = 0; nt < 4; ++nt) {
                const float pv = __expf(sc[nt][r] - mn);
                sc[nt][r] = pv;
                rs += pv;
            }
            rs += __shfl_xor(rs, 1);
            rs += __shfl_xor(rs, 2);
            rs += __shfl_xor(rs, 4);
            rs += __shfl_xor(rs, 8);
            lsum[r] = lsum[r] * alpha + rs;
            #pragma unroll
            for (int t = 0; t < 4; ++t) o[t][r] = o[t][r] * alpha;
        }
        #pragma unroll
        for (int r = 0; r < 8; ++r) {
            const int row = r + 8 * lhalf;
            #pragma unroll
            for (int nt = 0; nt < 4; ++nt)
                pw[row * LDK + nt * 16 + l16] = f2bf(sc[nt][r]);
        }
        __syncthreads();
        #pragma unroll
        for (int c = 0; c < 2; ++c) {
            FragBF pf;
            const unsigned short* pp = &pw[l16 * LDK + 32 * c + 16 * lhalf];
            pf.q[0] = *(const uint4*)pp;
            pf.q[1] = *(const uint4*)(pp + 8);
            #pragma unroll
            for (int vt = 0; vt < 4; ++vt) {
                FragBF vf;
                const unsigned short* vp = &sVT[(vt * 16 + l16) * LDK + 32 * c + 16 * lhalf];
                vf.q[0] = *(const uint4*)vp; vf.q[1] = *(const uint4*)(vp + 8);
                o[vt] = __builtin_amdgcn_wmma_f32_16x16x32_bf16(false, pf.v, false, vf.v,
                                                                (short)0, o[vt], false, false);
            }
        }
    }
    #pragma unroll
    for (int r = 0; r < 8; ++r) {
        const float inv = 1.0f / lsum[r];
        const int   row = r + 8 * lhalf;
        float* orow = Og + base_bh + (size_t)(q0 + row) * D_MODEL;
        #pragma unroll
        for (int vt = 0; vt < 4; ++vt)
            orow[vt * 16 + l16] = o[vt][r] * inv;
    }
}

// --------------------------- host launcher ---------------------------------
extern "C" void kernel_launch(void* const* d_in, const int* in_sizes, int n_in,
                              void* d_out, int out_size, void* d_ws, size_t ws_size,
                              hipStream_t stream) {
    const float* Q = (const float*)d_in[0];
    const float* K = (const float*)d_in[1];
    const float* V = (const float*)d_in[2];
    float*       O = (float*)d_out;

    const int    B    = in_sizes[0] / (S_LEN * D_MODEL);
    const size_t nElm = (size_t)B * S_LEN * D_MODEL;
    const size_t need = nElm * 2 /*bytes*/ * 2 /*K and V*/;

    dim3 grid(S_LEN / 64, N_HEADS, B);
    dim3 block(128);

    if (ws_size >= need) {
        unsigned short* Kbf = (unsigned short*)d_ws;
        unsigned short* Vtb = Kbf + nElm;
        convert_bf16_kernel<<<dim3((unsigned)(nElm / (256 * 8))), dim3(256), 0, stream>>>(K, Kbf);
        transpose_v_kernel <<<dim3(S_LEN / 64, N_HEADS, B), dim3(256), 0, stream>>>(V, Vtb);
        fattn_wmma_tdm<<<grid, block, 0, stream>>>(Q, Kbf, Vtb, O);
    } else {
        fattn_wmma_bf16_fb<<<grid, block, 0, stream>>>(Q, K, V, O);
    }
}
